// WindowCrossAttentionFusion_41266045780058
// MI455X (gfx1250) — matrix operand order repaired and checked by reading, not compiled, not run on hardware
//
#include <hip/hip_runtime.h>
#include <hip/hip_bf16.h>

typedef __attribute__((ext_vector_type(16))) _Float16 v16h;
typedef __attribute__((ext_vector_type(8)))  _Float16 v8h;
typedef __attribute__((ext_vector_type(8)))  float    v8f;
typedef __attribute__((ext_vector_type(4)))  float    v4f;

#define N_TOK  64
#define NHEADS 6
#define LF     64
#define HF     192
#define HD     32
#define HW     65536   // 256*256
#define IMG_W  256

// ---------------- WMMA fragment helpers (CDNA5 16x16x32 f16 layouts) ----------------

// A-matrix 16x32 f16: lane l -> row m0+(l&15); 16 consecutive K at k0+16*(l>=16)
__device__ __forceinline__ v16h frag_A(const _Float16* base, int rs, int m0, int k0, int lane) {
  const int row = m0 + (lane & 15);
  const int kb  = k0 + ((lane >> 4) << 4);
  const _Float16* p = base + row * rs + kb;
  v8h lo = *(const v8h*)p;
  v8h hi = *(const v8h*)(p + 8);
  return __builtin_shufflevector(lo, hi, 0,1,2,3,4,5,6,7,8,9,10,11,12,13,14,15);
}

// B-matrix 32x16 f16 stored [n][k]: lane l -> col n0+(l&15); 16 consecutive K at k0+16*(l>=16)
__device__ __forceinline__ v16h frag_B(const _Float16* base, int rs, int n0, int k0, int lane) {
  const int col = n0 + (lane & 15);
  const int kb  = k0 + ((lane >> 4) << 4);
  return *(const v16h*)(base + col * rs + kb);
}

__device__ __forceinline__ v8f wmma_f16(v16h a, v16h b, v8f c) {
  return __builtin_amdgcn_wmma_f32_16x16x32_f16(false, a, false, b, (short)0, c, false, false);
}

// ---------------- prep: transpose+fp16-convert weights, expand RPB bias ----------------

__global__ __launch_bounds__(256)
void wca_prep(const float* __restrict__ Wq, const float* __restrict__ Wk,
              const float* __restrict__ Wv, const float* __restrict__ Wo,
              const float* __restrict__ rpb,
              _Float16* __restrict__ WqT, _Float16* __restrict__ WkT,
              _Float16* __restrict__ WvT, _Float16* __restrict__ WoT,
              float* __restrict__ biasTab) {
  const int i = blockIdx.x * 256 + threadIdx.x;
  if (i < HF * LF) {                       // WqT[n][k] = Wq[k][n],  Wq is (LF,HF)
    const int n = i / LF, k = i % LF;
    WqT[i] = (_Float16)Wq[k * HF + n];
  }
  if (i < HF * HF) {                       // 192x192 transposes
    const int n = i / HF, k = i % HF;
    WkT[i] = (_Float16)Wk[k * HF + n];
    WvT[i] = (_Float16)Wv[k * HF + n];
    WoT[i] = (_Float16)Wo[k * HF + n];
  }
  if (i < NHEADS * N_TOK * N_TOK) {        // biasTab[h][q][k] = rpb[idx(q,k)][h]
    const int h = i / (N_TOK * N_TOK);
    const int r = i % (N_TOK * N_TOK);
    const int qi = r / N_TOK, kj = r % N_TOK;
    const int yi = qi >> 3, xi = qi & 7, yj = kj >> 3, xj = kj & 7;
    const int idx = (yi - yj + 7) * 15 + (xi - xj + 7);
    biasTab[i] = rpb[idx * NHEADS + h];
  }
}

// ---------------- main fused kernel: one workgroup (8 wave32) per 8x8 window ----------------

__global__ __launch_bounds__(256)
void wca_main(const float* __restrict__ y_lf, const float* __restrict__ y_hf,
              const float* __restrict__ g_lf, const float* __restrict__ b_lf,
              const float* __restrict__ g_hf, const float* __restrict__ b_hf,
              const float* __restrict__ bq, const float* __restrict__ bk,
              const float* __restrict__ bv, const float* __restrict__ bo,
              const _Float16* __restrict__ WqT, const _Float16* __restrict__ WkT,
              const _Float16* __restrict__ WvT, const _Float16* __restrict__ WoT,
              const float* __restrict__ biasTab, float* __restrict__ out)
{
  extern __shared__ _Float16 smem[];
  _Float16* sLF = smem;                    // 64*64
  _Float16* sHF = sLF + 64 * 64;           // 64*192  normalized hf
  _Float16* sQ  = sHF + 64 * HF;           // 64*192  (reused as O later)
  _Float16* sK  = sQ  + 64 * HF;           // 64*192  [token][dim] == B layout for Q@K^T
  _Float16* sVt = sK  + 64 * HF;           // 192*64  [dim][token] == B layout for P@V
  _Float16* sP  = sVt + HF * 64;           // 6*64*64 logits -> probabilities

  const int tid  = threadIdx.x;
  const int lane = tid & 31;
  const int wv   = tid >> 5;

  const int wid   = blockIdx.x;            // 4096 windows
  const int batch = wid >> 10;
  const int wy    = (wid >> 5) & 31;
  const int wx    = wid & 31;
  const int h0 = wy << 3, w0 = wx << 3;

  // Warm the (last-used) output-projection weights into near caches.
  __builtin_prefetch(WoT + tid * 144, 0, 1);   // global_prefetch_b8, covers 192*192 f16

  // ---- phase 0: gather window + LayerNorm, write f16 to LDS ----
  {
    const int t  = tid >> 2;               // token 0..63, 4 threads per token
    const int s  = tid & 3;
    const int ty = t >> 3, tx = t & 7;
    const int pix = (h0 + ty) * IMG_W + (w0 + tx);

    // hf: 48 contiguous channels per thread
    float v[48];
    float sum = 0.f, ss = 0.f;
    const int c0 = s * 48;
    const float* src = y_hf + (size_t)(batch * HF + c0) * HW + pix;
    #pragma unroll
    for (int j = 0; j < 48; ++j) {
      const float x = src[(size_t)j * HW];
      v[j] = x; sum += x; ss += x * x;
    }
    sum += __shfl_xor(sum, 1); sum += __shfl_xor(sum, 2);
    ss  += __shfl_xor(ss, 1);  ss  += __shfl_xor(ss, 2);
    float mu  = sum * (1.f / HF);
    float inv = rsqrtf(ss * (1.f / HF) - mu * mu + 1e-5f);
    #pragma unroll
    for (int j = 0; j < 48; ++j)
      sHF[t * HF + c0 + j] = (_Float16)((v[j] - mu) * inv * g_hf[c0 + j] + b_hf[c0 + j]);

    // lf: 16 contiguous channels per thread
    float u[16];
    sum = 0.f; ss = 0.f;
    const int c1 = s * 16;
    const float* srcl = y_lf + (size_t)(batch * LF + c1) * HW + pix;
    #pragma unroll
    for (int j = 0; j < 16; ++j) {
      const float x = srcl[(size_t)j * HW];
      u[j] = x; sum += x; ss += x * x;
    }
    sum += __shfl_xor(sum, 1); sum += __shfl_xor(sum, 2);
    ss  += __shfl_xor(ss, 1);  ss  += __shfl_xor(ss, 2);
    mu  = sum * (1.f / LF);
    inv = rsqrtf(ss * (1.f / LF) - mu * mu + 1e-5f);
    #pragma unroll
    for (int j = 0; j < 16; ++j)
      sLF[t * LF + c1 + j] = (_Float16)((u[j] - mu) * inv * g_lf[c1 + j] + b_lf[c1 + j]);
  }
  __syncthreads();

  const float qscale = 0.17677669529663689f;   // 1/sqrt(HD)

  // ---- phase 1: Q,K,V projections (4x12 tile grid each; 6 tiles/wave) ----
  // K and V share A fragments (same LN(hf) tile) -> one frag_A feeds two WMMAs.
  for (int i = 0; i < 6; ++i) {
    const int t  = wv + (i << 3);
    const int m0 = (t / 12) << 4;
    const int n0 = (t % 12) << 4;
    const int col = n0 + (lane & 15);
    const int r0  = m0 + ((lane >> 4) << 3);

    // Q = LN(lf) @ Wq, fold in bq and qscale
    v8f accq = {};
    #pragma unroll
    for (int ks = 0; ks < 2; ++ks)
      accq = wmma_f16(frag_A(sLF, LF, m0, ks * 32, lane),
                      frag_B(WqT, LF, n0, ks * 32, lane), accq);
    {
      const float bb = bq[col];
      #pragma unroll
      for (int vv = 0; vv < 8; ++vv)
        sQ[(r0 + vv) * HF + col] = (_Float16)((accq[vv] + bb) * qscale);
    }

    // K,V = LN(hf) @ {Wk,Wv} with shared A fragments
    v8f acck = {}, accv = {};
    #pragma unroll
    for (int ks = 0; ks < 6; ++ks) {
      v16h a = frag_A(sHF, HF, m0, ks * 32, lane);
      acck = wmma_f16(a, frag_B(WkT, HF, n0, ks * 32, lane), acck);
      accv = wmma_f16(a, frag_B(WvT, HF, n0, ks * 32, lane), accv);
    }
    {
      const float bbk = bk[col];
      const float bbv = bv[col];
      #pragma unroll
      for (int vv = 0; vv < 8; ++vv) {
        sK [(r0 + vv) * HF + col]    = (_Float16)(acck[vv] + bbk);
        sVt[col * N_TOK + r0 + vv]   = (_Float16)(accv[vv] + bbv);  // transposed store
      }
    }
  }
  __syncthreads();

  // ---- phase 2: S = (Q*scale) K^T + bias -> sP (f16), 96 tiles, K=HD=32 -> 1 WMMA each ----
  for (int i = 0; i < 12; ++i) {
    const int t  = wv + (i << 3);            // 0..95
    const int h  = t >> 4;
    const int r  = t & 15;
    const int m0 = (r >> 2) << 4;
    const int n0 = (r & 3) << 4;
    v8f acc = {};
    acc = wmma_f16(frag_A(sQ, HF, m0, h * HD, lane),
                   frag_B(sK, HF, n0, h * HD, lane), acc);
    const int col = n0 + (lane & 15);
    const int r0  = m0 + ((lane >> 4) << 3);
    const float* bt = biasTab + h * (N_TOK * N_TOK);
    _Float16*   dst = sP + h * (N_TOK * N_TOK);
    #pragma unroll
    for (int vv = 0; vv < 8; ++vv)
      dst[(r0 + vv) * N_TOK + col] = (_Float16)(acc[vv] + bt[(r0 + vv) * N_TOK + col]);
  }
  __syncthreads();

  // ---- phase 3: softmax over 6*64 rows of 64; vectorized b128 LDS traffic ----
  for (int r = tid; r < NHEADS * N_TOK; r += 256) {
    _Float16* prow = sP + r * N_TOK;
    v8h c[8];
    #pragma unroll
    for (int q = 0; q < 8; ++q) c[q] = *(const v8h*)(prow + 8 * q);
    float x[64];
    float mx = -3.0e38f;
    #pragma unroll
    for (int q = 0; q < 8; ++q)
      #pragma unroll
      for (int e = 0; e < 8; ++e) {
        const float f = (float)c[q][e];
        x[q * 8 + e] = f;
        mx = fmaxf(mx, f);
      }
    float ssum = 0.f;
    #pragma unroll
    for (int j = 0; j < 64; ++j) { x[j] = __expf(x[j] - mx); ssum += x[j]; }
    const float inv = 1.f / ssum;
    #pragma unroll
    for (int q = 0; q < 8; ++q) {
      v8h o;
      #pragma unroll
      for (int e = 0; e < 8; ++e) o[e] = (_Float16)(x[q * 8 + e] * inv);
      *(v8h*)(prow + 8 * q) = o;
    }
  }
  __syncthreads();

  // ---- phase 4: O = P @ V -> sO (reuses sQ), 48 tiles, K=64 -> 2 WMMA ----
  _Float16* sO = sQ;
  for (int i = 0; i < 6; ++i) {
    const int t  = wv + (i << 3);            // 0..47
    const int h  = t >> 3;
    const int r  = t & 7;
    const int m0 = (r >> 1) << 4;
    const int nd = (r & 1) << 4;
    v8f acc = {};
    #pragma unroll
    for (int ks = 0; ks < 2; ++ks)
      acc = wmma_f16(frag_A(sP + h * (N_TOK * N_TOK), N_TOK, m0, ks * 32, lane),
                     frag_B(sVt, N_TOK, h * HD + nd, ks * 32, lane), acc);
    const int col = h * HD + nd + (lane & 15);
    const int r0  = m0 + ((lane >> 4) << 3);
    #pragma unroll
    for (int vv = 0; vv < 8; ++vv)
      sO[(r0 + vv) * HF + col] = (_Float16)acc[vv];
  }
  __syncthreads();

  // ---- phase 5: Y = O @ Wo + bo + residual; coalesced 32B global RMW per lane/tile ----
  for (int i = 0; i < 6; ++i) {
    const int t  = wv + (i << 3);
    const int m0 = (t / 12) << 4;
    const int n0 = (t % 12) << 4;
    v8f acc = {};
    #pragma unroll
    for (int ks = 0; ks < 6; ++ks)
      acc = wmma_f16(frag_A(sO, HF, m0, ks * 32, lane),
                     frag_B(WoT, HF, n0, ks * 32, lane), acc);
    const int col = n0 + (lane & 15);
    const int r0  = m0 + ((lane >> 4) << 3);  // 8 consecutive tokens == one pixel row
    const size_t g = (size_t)(batch * HF + col) * HW
                   + (size_t)(h0 + (r0 >> 3)) * IMG_W + w0;
    const float bb = bo[col];
    v4f rlo = *(const v4f*)(y_hf + g);
    v4f rhi = *(const v4f*)(y_hf + g + 4);
    v4f olo, ohi;
    #pragma unroll
    for (int vv = 0; vv < 4; ++vv) {
      olo[vv] = acc[vv]     + bb + rlo[vv];
      ohi[vv] = acc[vv + 4] + bb + rhi[vv];
    }
    *(v4f*)(out + g)     = olo;
    *(v4f*)(out + g + 4) = ohi;
  }
}

// ---------------- launch ----------------

extern "C" void kernel_launch(void* const* d_in, const int* in_sizes, int n_in,
                              void* d_out, int out_size, void* d_ws, size_t ws_size,
                              hipStream_t stream) {
  const float* y_lf = (const float*)d_in[0];
  const float* y_hf = (const float*)d_in[1];
  const float* g_lf = (const float*)d_in[2];
  const float* b_lf = (const float*)d_in[3];
  const float* g_hf = (const float*)d_in[4];
  const float* b_hf = (const float*)d_in[5];
  const float* Wq   = (const float*)d_in[6];
  const float* bq   = (const float*)d_in[7];
  const float* Wk   = (const float*)d_in[8];
  const float* bk   = (const float*)d_in[9];
  const float* Wv   = (const float*)d_in[10];
  const float* bv   = (const float*)d_in[11];
  const float* Wo   = (const float*)d_in[12];
  const float* bo   = (const float*)d_in[13];
  const float* rpb  = (const float*)d_in[14];

  char* ws = (char*)d_ws;                 // layout (bytes):
  _Float16* WqT = (_Float16*)(ws + 0);         //  24576: [192][64]  f16
  _Float16* WkT = (_Float16*)(ws + 24576);     //  73728: [192][192] f16
  _Float16* WvT = (_Float16*)(ws + 98304);     //  73728
  _Float16* WoT = (_Float16*)(ws + 172032);    //  73728
  float* biasTab = (float*)(ws + 245760);      //  98304: [6][64][64] f32

  wca_prep<<<144, 256, 0, stream>>>(Wq, Wk, Wv, Wo, rpb, WqT, WkT, WvT, WoT, biasTab);

  const size_t lds_bytes = (size_t)(64*64 + 4*64*192 + 6*64*64) * sizeof(_Float16); // 155648
  (void)hipFuncSetAttribute((const void*)wca_main,
                            hipFuncAttributeMaxDynamicSharedMemorySize,
                            (int)lds_bytes);
  wca_main<<<4096, 256, lds_bytes, stream>>>(y_lf, y_hf, g_lf, b_lf, g_hf, b_hf,
                                             bq, bk, bv, bo,
                                             WqT, WkT, WvT, WoT, biasTab,
                                             (float*)d_out);
}
